// SelfAttention_18657337934540
// MI455X (gfx1250) — matrix operand order repaired and checked by reading, not compiled, not run on hardware
//
#include <hip/hip_runtime.h>
#include <stdint.h>

// ---------------------------------------------------------------------------
// Self-attention forward for MI455X (gfx1250, wave32, WMMA).
//   q = x Wq^T, k = x Wk^T, v = x Wv^T          (bf16 operands, f32 accum)
//   s = q k^T + (-1e9)*mask ; s = softmax(s)    (f32)
//   out = s v                                    (bf16 operands, f32 accum)
// Outputs concatenated: out [B,S,D] then s [B,S,S], both f32.
//
// GEMMs: 128x128x32 block tiles, 8 wave32s, each wave 64x32 (4x2 WMMA tiles).
// score_gemm / out_gemm stage bf16 operands with GLOBAL_LOAD_ASYNC_TO_LDS_B128
// (ASYNCcnt) double-buffered behind the WMMAs; fp32->bf16 paths use register
// double-buffering. One barrier per K-step.
// ---------------------------------------------------------------------------

typedef __attribute__((ext_vector_type(16))) __bf16 v16bf;
typedef __attribute__((ext_vector_type(8)))  float  v8f;

#define B_DIM 8
#define S_DIM 2048
#define D_DIM 1024
#define LDS_STRIDE 40            // 32 halves data + 16B pad; 80B rows, 16B aligned
#define TILE_USH (128 * LDS_STRIDE)
#define TILE_BYTES (TILE_USH * 2)

struct __align__(16) U128 { unsigned int x, y, z, w; };
struct __align__(16) F128 { float x, y, z, w; };
struct __align__(8)  U64_ { unsigned int x, y; };

union FragU {
  struct { U128 lo, hi; } u;
  v16bf v;
};

// async DMA: global (16B/lane) -> LDS, tracked by ASYNCcnt
#define ASYNC_B128(ldsoff, gaddr)                                         \
  asm volatile("global_load_async_to_lds_b128 %0, %1, off"                \
               :: "v"(ldsoff), "v"(gaddr) : "memory")
#define WAIT_ASYNC0() asm volatile("s_wait_asynccnt 0x0" ::: "memory")

static __device__ __forceinline__ unsigned lds_off32(const void* p) {
  return (unsigned)(uintptr_t)p;   // FLAT aperture: LDS offset = addr[31:0]
}

static __device__ __forceinline__ unsigned short f32_to_bf16(float f) {
  unsigned int u = __float_as_uint(f);
  u += 0x7FFFu + ((u >> 16) & 1u);   // round-to-nearest-even
  return (unsigned short)(u >> 16);
}

static __device__ __forceinline__ U64_ cvt4_bf16(F128 f) {
  U64_ r;
  r.x = (unsigned int)f32_to_bf16(f.x) | ((unsigned int)f32_to_bf16(f.y) << 16);
  r.y = (unsigned int)f32_to_bf16(f.z) | ((unsigned int)f32_to_bf16(f.w) << 16);
  return r;
}

// A-fragment (16x32 bf16, M x K): lane<16 -> row M=lane, K {0..7,16..23};
// lane>=16 -> row M=lane-16, K {8..15,24..31}.  Two b128 LDS reads.
static __device__ __forceinline__ v16bf load_a_frag(const unsigned short* tile,
                                                    int mrow, int l, int hi) {
  const U128* rp = (const U128*)(tile + (mrow + l) * LDS_STRIDE);
  FragU f;
  f.u.lo = rp[hi];
  f.u.hi = rp[2 + hi];
  return f.v;
}

// B-fragment (32x16 bf16, K x N) staged K-contiguous per N row:
// lane<16 -> N=lane, K {0..7,16..23} ; lane>=16 -> N=lane-16, K {8..15,24..31}.
static __device__ __forceinline__ v16bf load_b_frag(const unsigned short* tile,
                                                    int nrow, int l, int hi) {
  const U128* rp = (const U128*)(tile + (nrow + l) * LDS_STRIDE);
  FragU f;
  f.u.lo = rp[hi];
  f.u.hi = rp[2 + hi];
  return f.v;
}

// ---------------------------------------------------------------------------
// Kernel 1: projections. C[M=B*S, N=D] = X[M,K] * W[N,K]^T, K=D.
// blockIdx.z selects Wq/Wk/Wv. q,k stored row-major bf16 [B*S][D];
// v stored transposed bf16 [B][D][S] so PV GEMM's B operand is K-contiguous.
// fp32->bf16 conversion on the staging path -> register double buffering.
// ---------------------------------------------------------------------------
__global__ __launch_bounds__(256) void proj_gemm(
    const float* __restrict__ X,
    const float* __restrict__ Wq, const float* __restrict__ Wk,
    const float* __restrict__ Wv,
    unsigned short* __restrict__ Qb, unsigned short* __restrict__ Kb,
    unsigned short* __restrict__ VbT) {
  __shared__ unsigned short As[2 * TILE_USH];
  __shared__ unsigned short Bs[2 * TILE_USH];

  const int tid  = threadIdx.x;
  const int lane = tid & 31, w = tid >> 5;
  const int l = lane & 15, hi = lane >> 4;
  const int wm = w & 1, wn = w >> 1;          // 2 waves in M, 4 in N
  const int m0 = blockIdx.x * 128;
  const int n0 = blockIdx.y * 128;
  const int proj = blockIdx.z;
  const float* W = (proj == 0) ? Wq : (proj == 1) ? Wk : Wv;
  const int K = D_DIM;

  v8f acc[4][2];
  const v8f vzero = {0.f, 0.f, 0.f, 0.f, 0.f, 0.f, 0.f, 0.f};
#pragma unroll
  for (int i = 0; i < 4; ++i)
#pragma unroll
    for (int j = 0; j < 2; ++j) acc[i][j] = vzero;

  U64_ aR[4], bR[4];
#pragma unroll
  for (int i = 0; i < 4; ++i) {                 // preload K-tile 0
    int c = tid + i * 256, row = c >> 3, col = (c & 7) << 2;
    aR[i] = cvt4_bf16(*(const F128*)&X[(size_t)(m0 + row) * K + col]);
    bR[i] = cvt4_bf16(*(const F128*)&W[(size_t)(n0 + row) * K + col]);
  }

  int buf = 0;
  for (int k0 = 0; k0 < K; k0 += 32) {
#pragma unroll
    for (int i = 0; i < 4; ++i) {
      int c = tid + i * 256, row = c >> 3, col = (c & 7) << 2;
      int o = buf * TILE_USH + row * LDS_STRIDE + col;
      *(U64_*)&As[o] = aR[i];
      *(U64_*)&Bs[o] = bR[i];
    }
    __syncthreads();

    if (k0 + 32 < K) {
      int k = k0 + 32;
#pragma unroll
      for (int i = 0; i < 4; ++i) {
        int c = tid + i * 256, row = c >> 3, col = (c & 7) << 2;
        aR[i] = cvt4_bf16(*(const F128*)&X[(size_t)(m0 + row) * K + k + col]);
        bR[i] = cvt4_bf16(*(const F128*)&W[(size_t)(n0 + row) * K + k + col]);
      }
      if (k0 + 64 < K) {   // warm L2 for the tile after next
        int row = tid >> 3, col = (tid & 7) << 2;
        __builtin_prefetch(&X[(size_t)(m0 + row) * K + k0 + 64 + col], 0, 0);
        __builtin_prefetch(&W[(size_t)(n0 + row) * K + k0 + 64 + col], 0, 0);
      }
    }

    const unsigned short* At = As + buf * TILE_USH;
    const unsigned short* Bt = Bs + buf * TILE_USH;
    v16bf af[4], bf[2];
#pragma unroll
    for (int i = 0; i < 4; ++i) af[i] = load_a_frag(At, wm * 64 + i * 16, l, hi);
#pragma unroll
    for (int j = 0; j < 2; ++j) bf[j] = load_b_frag(Bt, wn * 32 + j * 16, l, hi);
#pragma unroll
    for (int i = 0; i < 4; ++i)
#pragma unroll
      for (int j = 0; j < 2; ++j)
        acc[i][j] = __builtin_amdgcn_wmma_f32_16x16x32_bf16(
            false, af[i], false, bf[j], (short)0, acc[i][j], false, false);
    buf ^= 1;
  }

  // epilogue: proj is block-uniform -> branch hoisted out of store loops
  if (proj < 2) {
    unsigned short* Out = (proj == 0) ? Qb : Kb;
#pragma unroll
    for (int i = 0; i < 4; ++i)
#pragma unroll
      for (int j = 0; j < 2; ++j) {
        int tm = m0 + wm * 64 + i * 16;
        int tn = n0 + wn * 32 + j * 16 + l;
#pragma unroll
        for (int r = 0; r < 8; ++r)
          Out[(size_t)(tm + r + hi * 8) * D_DIM + tn] = f32_to_bf16(acc[i][j][r]);
      }
  } else {
#pragma unroll
    for (int i = 0; i < 4; ++i)
#pragma unroll
      for (int j = 0; j < 2; ++j) {
        int tm = m0 + wm * 64 + i * 16;
        int tn = n0 + wn * 32 + j * 16 + l;
#pragma unroll
        for (int r = 0; r < 8; ++r) {
          int gm = tm + r + hi * 8;
          int b = gm >> 11, si = gm & (S_DIM - 1);
          VbT[((size_t)b * D_DIM + tn) * S_DIM + si] = f32_to_bf16(acc[i][j][r]);
        }
      }
  }
}

// ---------------------------------------------------------------------------
// Kernel 2: scores. S[q,kk] = sum_d Q[q,d]*K[kk,d], + (-1e9)*mask[kk]. f32 out.
// Both operands already bf16 -> fully async (DMA) double-buffered staging.
// ---------------------------------------------------------------------------
__global__ __launch_bounds__(256) void score_gemm(
    const unsigned short* __restrict__ Qb, const unsigned short* __restrict__ Kb,
    const int* __restrict__ mask, float* __restrict__ Sout) {
  __shared__ unsigned short As[2 * TILE_USH];
  __shared__ unsigned short Bs[2 * TILE_USH];

  const int tid  = threadIdx.x;
  const int lane = tid & 31, w = tid >> 5;
  const int l = lane & 15, hi = lane >> 4;
  const int wm = w & 1, wn = w >> 1;
  const int m0 = blockIdx.x * 128;
  const int n0 = blockIdx.y * 128;
  const int b  = blockIdx.z;
  const unsigned short* Q  = Qb + (size_t)b * S_DIM * D_DIM;
  const unsigned short* Kp = Kb + (size_t)b * S_DIM * D_DIM;
  float* Sp = Sout + (size_t)b * S_DIM * S_DIM;
  const int* mp = mask + (size_t)b * S_DIM;
  const int K = D_DIM;

  const unsigned ldsA = lds_off32(As);
  const unsigned ldsB = lds_off32(Bs);
  // this thread's two 16B chunks per tile
  const int c0 = tid,        r0 = c0 >> 2, q0 = (c0 & 3) << 3;
  const int c1 = tid + 256,  r1 = c1 >> 2, q1 = (c1 & 3) << 3;
  const unsigned lo0 = (unsigned)(r0 * LDS_STRIDE + q0) * 2;
  const unsigned lo1 = (unsigned)(r1 * LDS_STRIDE + q1) * 2;

  v8f acc[4][2];
  const v8f vzero = {0.f, 0.f, 0.f, 0.f, 0.f, 0.f, 0.f, 0.f};
#pragma unroll
  for (int i = 0; i < 4; ++i)
#pragma unroll
    for (int j = 0; j < 2; ++j) acc[i][j] = vzero;

  // kick off tile 0 -> buffer 0
  {
    ASYNC_B128(ldsA + lo0, (unsigned long long)(uintptr_t)&Q [(size_t)(m0 + r0) * K + q0]);
    ASYNC_B128(ldsA + lo1, (unsigned long long)(uintptr_t)&Q [(size_t)(m0 + r1) * K + q1]);
    ASYNC_B128(ldsB + lo0, (unsigned long long)(uintptr_t)&Kp[(size_t)(n0 + r0) * K + q0]);
    ASYNC_B128(ldsB + lo1, (unsigned long long)(uintptr_t)&Kp[(size_t)(n0 + r1) * K + q1]);
  }

  int buf = 0;
  for (int k0 = 0; k0 < K; k0 += 32) {
    WAIT_ASYNC0();        // own DMAs done...
    __syncthreads();      // ...then everyone's are visible
    if (k0 + 32 < K) {    // DMA next tile into the other buffer, behind WMMAs
      int k = k0 + 32;
      unsigned base = (unsigned)((buf ^ 1) * TILE_BYTES);
      ASYNC_B128(ldsA + base + lo0, (unsigned long long)(uintptr_t)&Q [(size_t)(m0 + r0) * K + k + q0]);
      ASYNC_B128(ldsA + base + lo1, (unsigned long long)(uintptr_t)&Q [(size_t)(m0 + r1) * K + k + q1]);
      ASYNC_B128(ldsB + base + lo0, (unsigned long long)(uintptr_t)&Kp[(size_t)(n0 + r0) * K + k + q0]);
      ASYNC_B128(ldsB + base + lo1, (unsigned long long)(uintptr_t)&Kp[(size_t)(n0 + r1) * K + k + q1]);
      if (k0 + 64 < K) {
        __builtin_prefetch(&Q [(size_t)(m0 + r0) * K + k0 + 64 + q0], 0, 0);
        __builtin_prefetch(&Kp[(size_t)(n0 + r0) * K + k0 + 64 + q0], 0, 0);
      }
    }

    const unsigned short* At = As + buf * TILE_USH;
    const unsigned short* Bt = Bs + buf * TILE_USH;
    v16bf af[4], bf[2];
#pragma unroll
    for (int i = 0; i < 4; ++i) af[i] = load_a_frag(At, wm * 64 + i * 16, l, hi);
#pragma unroll
    for (int j = 0; j < 2; ++j) bf[j] = load_b_frag(Bt, wn * 32 + j * 16, l, hi);
#pragma unroll
    for (int i = 0; i < 4; ++i)
#pragma unroll
      for (int j = 0; j < 2; ++j)
        acc[i][j] = __builtin_amdgcn_wmma_f32_16x16x32_bf16(
            false, af[i], false, bf[j], (short)0, acc[i][j], false, false);
    buf ^= 1;
  }

#pragma unroll
  for (int i = 0; i < 4; ++i)
#pragma unroll
    for (int j = 0; j < 2; ++j) {
      int tm = m0 + wm * 64 + i * 16;
      int tn = n0 + wn * 32 + j * 16 + l;
      float madd = -1.0e9f * (float)mp[tn];   // mask broadcast over queries
#pragma unroll
      for (int r = 0; r < 8; ++r)
        Sp[(size_t)(tm + r + hi * 8) * S_DIM + tn] = acc[i][j][r] + madd;
    }
}

// ---------------------------------------------------------------------------
// Kernel 3: row softmax in place, one 256-thread block per row of 2048.
// ---------------------------------------------------------------------------
__global__ __launch_bounds__(256) void softmax_rows(float* __restrict__ Sreg) {
  __shared__ float red[256];
  const int tid = threadIdx.x;
  float* row = Sreg + (size_t)blockIdx.x * S_DIM;

  float vals[8];
  float m = -3.0e38f;
#pragma unroll
  for (int i = 0; i < 8; ++i) {
    vals[i] = row[tid + i * 256];
    m = fmaxf(m, vals[i]);
  }
  red[tid] = m;
  __syncthreads();
  for (int s = 128; s > 0; s >>= 1) {
    if (tid < s) red[tid] = fmaxf(red[tid], red[tid + s]);
    __syncthreads();
  }
  float rmax = red[0];
  __syncthreads();

  float sum = 0.f;
#pragma unroll
  for (int i = 0; i < 8; ++i) {
    vals[i] = __expf(vals[i] - rmax);
    sum += vals[i];
  }
  red[tid] = sum;
  __syncthreads();
  for (int s = 128; s > 0; s >>= 1) {
    if (tid < s) red[tid] += red[tid + s];
    __syncthreads();
  }
  float inv = 1.0f / red[0];
#pragma unroll
  for (int i = 0; i < 8; ++i) row[tid + i * 256] = vals[i] * inv;
}

// ---------------------------------------------------------------------------
// Kernel 4: out[q,d] = sum_k P[q,k] * V[k,d].
// A (probs): fp32->bf16 register staging.  B (V^T tile): async DMA staging.
// ---------------------------------------------------------------------------
__global__ __launch_bounds__(256) void out_gemm(
    const float* __restrict__ Sreg, const unsigned short* __restrict__ VbT,
    float* __restrict__ Out) {
  __shared__ unsigned short As[2 * TILE_USH];
  __shared__ unsigned short Bs[2 * TILE_USH];

  const int tid  = threadIdx.x;
  const int lane = tid & 31, w = tid >> 5;
  const int l = lane & 15, hi = lane >> 4;
  const int wm = w & 1, wn = w >> 1;
  const int m0 = blockIdx.x * 128;
  const int n0 = blockIdx.y * 128;
  const int b  = blockIdx.z;
  const float* Sp = Sreg + (size_t)b * S_DIM * S_DIM;
  const unsigned short* Vp = VbT + (size_t)b * D_DIM * S_DIM;
  float* Op = Out + (size_t)b * S_DIM * D_DIM;
  const int K = S_DIM;

  const unsigned ldsB = lds_off32(Bs);
  const int c0 = tid,       r0 = c0 >> 2, q0 = (c0 & 3) << 3;
  const int c1 = tid + 256, r1 = c1 >> 2, q1 = (c1 & 3) << 3;
  const unsigned lo0 = (unsigned)(r0 * LDS_STRIDE + q0) * 2;
  const unsigned lo1 = (unsigned)(r1 * LDS_STRIDE + q1) * 2;

  v8f acc[4][2];
  const v8f vzero = {0.f, 0.f, 0.f, 0.f, 0.f, 0.f, 0.f, 0.f};
#pragma unroll
  for (int i = 0; i < 4; ++i)
#pragma unroll
    for (int j = 0; j < 2; ++j) acc[i][j] = vzero;

  U64_ aR[4];   // fp32 probs converted to packed bf16
#pragma unroll
  for (int i = 0; i < 4; ++i) {
    int c = tid + i * 256, row = c >> 3, col = (c & 7) << 2;
    aR[i] = cvt4_bf16(*(const F128*)&Sp[(size_t)(m0 + row) * K + col]);
  }
  // kick off V tile 0 -> buffer 0 via DMA
  ASYNC_B128(ldsB + lo0, (unsigned long long)(uintptr_t)&Vp[(size_t)(n0 + r0) * S_DIM + q0]);
  ASYNC_B128(ldsB + lo1, (unsigned long long)(uintptr_t)&Vp[(size_t)(n0 + r1) * S_DIM + q1]);

  int buf = 0;
  for (int k0 = 0; k0 < K; k0 += 32) {
#pragma unroll
    for (int i = 0; i < 4; ++i) {
      int c = tid + i * 256, row = c >> 3, col = (c & 7) << 2;
      *(U64_*)&As[buf * TILE_USH + row * LDS_STRIDE + col] = aR[i];
    }
    WAIT_ASYNC0();
    __syncthreads();

    if (k0 + 32 < K) {
      int k = k0 + 32;
      unsigned base = (unsigned)((buf ^ 1) * TILE_BYTES);
      ASYNC_B128(ldsB + base + lo0, (unsigned long long)(uintptr_t)&Vp[(size_t)(n0 + r0) * S_DIM + k + q0]);
      ASYNC_B128(ldsB + base + lo1, (unsigned long long)(uintptr_t)&Vp[(size_t)(n0 + r1) * S_DIM + k + q1]);
#pragma unroll
      for (int i = 0; i < 4; ++i) {
        int c = tid + i * 256, row = c >> 3, col = (c & 7) << 2;
        aR[i] = cvt4_bf16(*(const F128*)&Sp[(size_t)(m0 + row) * K + k + col]);
      }
      if (k0 + 64 < K) {
        int row = tid >> 3, col = (tid & 7) << 2;
        __builtin_prefetch(&Sp[(size_t)(m0 + row) * K + k0 + 64 + col], 0, 0);
        __builtin_prefetch(&Vp[(size_t)(n0 + r0) * S_DIM + k0 + 64 + q0], 0, 0);
      }
    }

    const unsigned short* At = As + buf * TILE_USH;
    const unsigned short* Bt = Bs + buf * TILE_USH;
    v16bf af[4], bf[2];
#pragma unroll
    for (int i = 0; i < 4; ++i) af[i] = load_a_frag(At, wm * 64 + i * 16, l, hi);
#pragma unroll
    for (int j = 0; j < 2; ++j) bf[j] = load_b_frag(Bt, wn * 32 + j * 16, l, hi);
#pragma unroll
    for (int i = 0; i < 4; ++i)
#pragma unroll
      for (int j = 0; j < 2; ++j)
        acc[i][j] = __builtin_amdgcn_wmma_f32_16x16x32_bf16(
            false, af[i], false, bf[j], (short)0, acc[i][j], false, false);
    buf ^= 1;
  }

#pragma unroll
  for (int i = 0; i < 4; ++i)
#pragma unroll
    for (int j = 0; j < 2; ++j) {
      int tm = m0 + wm * 64 + i * 16;
      int tn = n0 + wn * 32 + j * 16 + l;
#pragma unroll
      for (int r = 0; r < 8; ++r)
        Op[(size_t)(tm + r + hi * 8) * D_DIM + tn] = acc[i][j][r];
    }
}

// ---------------------------------------------------------------------------
extern "C" void kernel_launch(void* const* d_in, const int* in_sizes, int n_in,
                              void* d_out, int out_size, void* d_ws, size_t ws_size,
                              hipStream_t stream) {
  const float* x    = (const float*)d_in[0];  // [B,S,D]
  const int*   mask = (const int*)d_in[1];    // [B,S]
  const float* wq   = (const float*)d_in[2];  // [D,D]
  const float* wk   = (const float*)d_in[3];
  const float* wv   = (const float*)d_in[4];

  float* out   = (float*)d_out;                         // [B,S,D]
  float* sattn = out + (size_t)B_DIM * S_DIM * D_DIM;   // [B,S,S]

  const size_t proj_elems = (size_t)B_DIM * S_DIM * D_DIM;  // bf16 each
  unsigned short* qb = (unsigned short*)d_ws;
  unsigned short* kb = qb + proj_elems;
  unsigned short* vt = kb + proj_elems;                     // [B][D][S]

  // 1) q,k,v projections (bf16 WMMA, f32 accum); v stored transposed
  proj_gemm<<<dim3((B_DIM * S_DIM) / 128, D_DIM / 128, 3), 256, 0, stream>>>(
      x, wq, wk, wv, qb, kb, vt);
  // 2) scores + mask -> s region of d_out (f32)
  score_gemm<<<dim3(S_DIM / 128, S_DIM / 128, B_DIM), 256, 0, stream>>>(
      qb, kb, mask, sattn);
  // 3) softmax rows in place
  softmax_rows<<<dim3(B_DIM * S_DIM), 256, 0, stream>>>(sattn);
  // 4) out = P @ V
  out_gemm<<<dim3(S_DIM / 128, D_DIM / 128, B_DIM), 256, 0, stream>>>(
      sattn, vt, out);
}
